// TransformerBlock_32461362823724
// MI455X (gfx1250) — compile-verified, matrix operand order checked by reading, not run on hardware
//
#include <hip/hip_runtime.h>
#include <hip/hip_bf16.h>
#include <math.h>

typedef __attribute__((ext_vector_type(16))) _Float16 v16h;
typedef __attribute__((ext_vector_type(8)))  float    v8f;

#define WAVES 8
#define Cdim 32
#define Hdim 64

// ---------- helpers ----------

__device__ __forceinline__ void wave_sync_lds() {
  // per-wave LDS ordering: DS ops are in-order within a wave; the wait +
  // memory clobber prevents compiler reordering across the staging tiles.
  asm volatile("s_wait_dscnt 0x0" ::: "memory");
}

// 16-bit A/B fragment element->K mapping (ISA 7.12.2, 16-bit 16x32):
// lane L: g=L/16, row=L%16; half t in v16h: k = 8g + t + (t>=8 ? 8 : 0)
__device__ __forceinline__ int frag_k(int t, int g) {
  return 8 * g + t + ((t >= 8) ? 8 : 0);
}

__device__ __forceinline__ v8f wmma_f16(v16h a, v16h b, v8f c) {
  return __builtin_amdgcn_wmma_f32_16x16x32_f16(false, a, false, b, (short)0, c,
                                                false, false);
}

// Build a B fragment (K x 16 slice) from a row-major [Krows x Ncols] f32
// weight matrix; K zero-padded beyond Krows.
__device__ __forceinline__ v16h load_bfrag(const float* W, int Krows, int Ncols,
                                           int kbase, int nbase, int lane) {
  int n = nbase + (lane & 15);
  int g = lane >> 4;
  v16h b;
#pragma unroll
  for (int t = 0; t < 16; ++t) {
    int k = kbase + frag_k(t, g);
    float val = (k < Krows) ? W[k * Ncols + n] : 0.f;
    b[t] = (_Float16)val;
  }
  return b;
}

// Build an A fragment (16 x 32 K-slice) from an LDS f32 tile [16 x stride]
__device__ __forceinline__ v16h afrag_lds(const float* buf, int stride,
                                          int kbase, int lane) {
  int m = lane & 15, g = lane >> 4;
  v16h a;
#pragma unroll
  for (int t = 0; t < 16; ++t)
    a[t] = (_Float16)buf[m * stride + kbase + frag_k(t, g)];
  return a;
}

// order-preserving float <-> uint map for atomicMax-based segment_max
__device__ __forceinline__ unsigned enc_f32(float f) {
  unsigned u = __float_as_uint(f);
  return (u & 0x80000000u) ? ~u : (u | 0x80000000u);
}
__device__ __forceinline__ float dec_f32(unsigned u) {
  return (u & 0x80000000u) ? __uint_as_float(u & 0x7FFFFFFFu)
                           : __uint_as_float(~u);
}

// ---------- init ----------

__global__ void init_bufs(unsigned* __restrict__ mb, float* __restrict__ denom,
                          float* __restrict__ acc, int n) {
  int i = blockIdx.x * blockDim.x + threadIdx.x;
  if (i < n) {
    mb[i] = 0x007FFFFFu;  // enc(-inf)
    denom[i] = 0.f;
    acc[i] = 0.f;
  }
}

// ---------- node-side GEMMs: h = relu(x@Win+b); v,a_src,a_dst = h@W* ----------

__device__ __forceinline__ void proj_store(v16h ah, v16h f0, v16h f1,
                                           float* __restrict__ outp, int base,
                                           int N, int lane) {
  int m = lane & 15, g = lane >> 4;
  v8f c0 = {};
  c0 = wmma_f16(ah, f0, c0);
#pragma unroll
  for (int r = 0; r < 8; ++r) {
    int rr = base + r + 8 * g;
    if (rr < N) outp[rr * Cdim + m] = c0[r];
  }
  v8f c1 = {};
  c1 = wmma_f16(ah, f1, c1);
#pragma unroll
  for (int r = 0; r < 8; ++r) {
    int rr = base + r + 8 * g;
    if (rr < N) outp[rr * Cdim + 16 + m] = c1[r];
  }
}

__global__ void node_in_kernel(const float* __restrict__ x, const float* Win,
                               const float* bin, const float* Wv,
                               const float* Wsrc, const float* Wdst,
                               float* __restrict__ vout,
                               float* __restrict__ asrc,
                               float* __restrict__ adst, int N) {
  __shared__ float s_h[WAVES][16 * Cdim];
  const int lane = threadIdx.x & 31, wave = threadIdx.x >> 5;
  const int m = lane & 15, g = lane >> 4;
  float* hbuf = s_h[wave];

  v16h win0 = load_bfrag(Win, Cdim, Cdim, 0, 0, lane);
  v16h win1 = load_bfrag(Win, Cdim, Cdim, 0, 16, lane);
  v16h wv0 = load_bfrag(Wv, Cdim, Cdim, 0, 0, lane);
  v16h wv1 = load_bfrag(Wv, Cdim, Cdim, 0, 16, lane);
  v16h ws0 = load_bfrag(Wsrc, Cdim, Cdim, 0, 0, lane);
  v16h ws1 = load_bfrag(Wsrc, Cdim, Cdim, 0, 16, lane);
  v16h wd0 = load_bfrag(Wdst, Cdim, Cdim, 0, 0, lane);
  v16h wd1 = load_bfrag(Wdst, Cdim, Cdim, 0, 16, lane);
  float b0 = bin[m], b1 = bin[16 + m];

  int nTiles = (N + 15) >> 4;
  for (int tile = blockIdx.x * WAVES + wave; tile < nTiles;
       tile += gridDim.x * WAVES) {
    int base = tile << 4;
    int row = base + m;
    if (row > N - 1) row = N - 1;
    v16h ax;
#pragma unroll
    for (int t = 0; t < 16; ++t)
      ax[t] = (_Float16)x[row * Cdim + frag_k(t, g)];

    v8f c0 = {};
    c0 = wmma_f16(ax, win0, c0);
#pragma unroll
    for (int r = 0; r < 8; ++r)
      hbuf[(r + 8 * g) * Cdim + m] = fmaxf(c0[r] + b0, 0.f);
    v8f c1 = {};
    c1 = wmma_f16(ax, win1, c1);
#pragma unroll
    for (int r = 0; r < 8; ++r)
      hbuf[(r + 8 * g) * Cdim + 16 + m] = fmaxf(c1[r] + b1, 0.f);
    wave_sync_lds();

    v16h ah = afrag_lds(hbuf, Cdim, 0, lane);
    proj_store(ah, wv0, wv1, vout, base, N, lane);
    proj_store(ah, ws0, ws1, asrc, base, N, lane);
    proj_store(ah, wd0, wd1, adst, base, N, lane);
    wave_sync_lds();
  }
}

// ---------- edge passes: recompute delta/alpha via WMMA each pass ----------
// MODE 0: alpha -> atomicMax m ; MODE 1: exp -> atomicAdd denom ;
// MODE 2: attn*(v+delta) -> atomicAdd acc

template <int MODE>
__global__ void edge_pass(const int* __restrict__ ei,
                          const float* __restrict__ pos,
                          const float* __restrict__ asrc,
                          const float* __restrict__ adst,
                          const float* __restrict__ vbuf, const float* pW1,
                          const float* pb1, const float* pW2, const float* pb2,
                          const float* aW1, const float* ab1, const float* aW2,
                          const float* ab2, unsigned* __restrict__ mbuf,
                          float* __restrict__ denom, float* __restrict__ acc,
                          int E) {
  __shared__ float s_h[WAVES][16 * Hdim];   // hidden (64) staging
  __shared__ float s_d[WAVES][16 * Cdim];   // delta staging
  __shared__ int s_si[WAVES][16];
  __shared__ int s_di[WAVES][16];

  const int lane = threadIdx.x & 31, wave = threadIdx.x >> 5;
  const int m = lane & 15, g = lane >> 4;
  float* hbuf = s_h[wave];
  float* dbuf = s_d[wave];
  int* si = s_si[wave];
  int* di = s_di[wave];

  // resident weight B-fragments (K zero-padded to 32/64)
  v16h pw1[4], aw1[4], pw2_0[2], pw2_1[2], aw2_0[2], aw2_1[2];
#pragma unroll
  for (int nt = 0; nt < 4; ++nt) {
    pw1[nt] = load_bfrag(pW1, 3, Hdim, 0, nt * 16, lane);
    aw1[nt] = load_bfrag(aW1, Cdim, Hdim, 0, nt * 16, lane);
  }
#pragma unroll
  for (int nt = 0; nt < 2; ++nt) {
    pw2_0[nt] = load_bfrag(pW2, Hdim, Cdim, 0, nt * 16, lane);
    pw2_1[nt] = load_bfrag(pW2, Hdim, Cdim, 32, nt * 16, lane);
    aw2_0[nt] = load_bfrag(aW2, Hdim, Cdim, 0, nt * 16, lane);
    aw2_1[nt] = load_bfrag(aW2, Hdim, Cdim, 32, nt * 16, lane);
  }
  float pb1v[4], ab1v[4], pb2v[2], ab2v[2];
#pragma unroll
  for (int nt = 0; nt < 4; ++nt) {
    pb1v[nt] = pb1[nt * 16 + m];
    ab1v[nt] = ab1[nt * 16 + m];
  }
#pragma unroll
  for (int nt = 0; nt < 2; ++nt) {
    pb2v[nt] = pb2[nt * 16 + m];
    ab2v[nt] = ab2[nt * 16 + m];
  }

  int nTiles = (E + 15) >> 4;
  for (int tile = blockIdx.x * WAVES + wave; tile < nTiles;
       tile += gridDim.x * WAVES) {
    int base = tile << 4;
    if (lane < 16) {
      int e = base + lane;
      if (e > E - 1) e = E - 1;
      si[lane] = ei[e];       // src row of edge_index
      di[lane] = ei[E + e];   // dst row
    }
    wave_sync_lds();

    int s = si[m], d = di[m];

    // --- pos_nn layer 1: [16x3 pad 32] @ [32x64] ---
    float pd0 = pos[d * 3 + 0] - pos[s * 3 + 0];
    float pd1 = pos[d * 3 + 1] - pos[s * 3 + 1];
    float pd2 = pos[d * 3 + 2] - pos[s * 3 + 2];
    v16h a1;
#pragma unroll
    for (int t = 0; t < 16; ++t) {
      int k = frag_k(t, g);
      float val = (k == 0) ? pd0 : (k == 1) ? pd1 : (k == 2) ? pd2 : 0.f;
      a1[t] = (_Float16)val;
    }
#pragma unroll
    for (int nt = 0; nt < 4; ++nt) {
      v8f c = {};
      c = wmma_f16(a1, pw1[nt], c);
#pragma unroll
      for (int r = 0; r < 8; ++r)
        hbuf[(r + 8 * g) * Hdim + nt * 16 + m] = fmaxf(c[r] + pb1v[nt], 0.f);
    }
    wave_sync_lds();

    // --- pos_nn layer 2: [16x64] @ [64x32] -> delta in dbuf ---
#pragma unroll
    for (int nt = 0; nt < 2; ++nt) {
      v8f c = {};
      c = wmma_f16(afrag_lds(hbuf, Hdim, 0, lane), pw2_0[nt], c);
      c = wmma_f16(afrag_lds(hbuf, Hdim, 32, lane), pw2_1[nt], c);
#pragma unroll
      for (int r = 0; r < 8; ++r)
        dbuf[(r + 8 * g) * Cdim + nt * 16 + m] = fmaxf(c[r] + pb2v[nt], 0.f);
    }
    wave_sync_lds();

    // --- attn input Y = a_dst[d] - a_src[s] + delta (L2-resident gathers) ---
    v16h ay;
#pragma unroll
    for (int t = 0; t < 16; ++t) {
      int k = frag_k(t, g);
      float val = adst[d * Cdim + k] - asrc[s * Cdim + k] + dbuf[m * Cdim + k];
      ay[t] = (_Float16)val;
    }

    // --- attn_nn layer 1: [16x32] @ [32x64] ---
#pragma unroll
    for (int nt = 0; nt < 4; ++nt) {
      v8f c = {};
      c = wmma_f16(ay, aw1[nt], c);
#pragma unroll
      for (int r = 0; r < 8; ++r)
        hbuf[(r + 8 * g) * Hdim + nt * 16 + m] = fmaxf(c[r] + ab1v[nt], 0.f);
    }
    wave_sync_lds();

    // --- attn_nn layer 2: [16x64] @ [64x32] -> alpha, consumed in-place ---
#pragma unroll
    for (int nt = 0; nt < 2; ++nt) {
      v8f c = {};
      c = wmma_f16(afrag_lds(hbuf, Hdim, 0, lane), aw2_0[nt], c);
      c = wmma_f16(afrag_lds(hbuf, Hdim, 32, lane), aw2_1[nt], c);
#pragma unroll
      for (int r = 0; r < 8; ++r) {
        float al = fmaxf(c[r] + ab2v[nt], 0.f);
        int mr = r + 8 * g;
        int e = base + mr;
        if (e < E) {
          int dd = di[mr];
          int ch = nt * 16 + m;
          int idx = dd * Cdim + ch;
          if (MODE == 0) {
            atomicMax(&mbuf[idx], enc_f32(al));
          } else {
            float mx = dec_f32(mbuf[idx]);
            float ex = __expf(al - mx);
            if (MODE == 1) {
              atomicAdd(&denom[idx], ex);
            } else {
              float attn = ex / (denom[idx] + 1e-16f);
              int ss = si[mr];
              float msg = attn * (vbuf[ss * Cdim + ch] + dbuf[mr * Cdim + ch]);
              atomicAdd(&acc[idx], msg);
            }
          }
        }
      }
    }
    wave_sync_lds();
  }
}

// ---------- final: out = relu(acc @ Wout + bout) ----------

__global__ void node_out_kernel(const float* __restrict__ acc,
                                const float* Wout, const float* bout,
                                float* __restrict__ out, int N) {
  const int lane = threadIdx.x & 31, wave = threadIdx.x >> 5;
  const int m = lane & 15, g = lane >> 4;
  v16h w0 = load_bfrag(Wout, Cdim, Cdim, 0, 0, lane);
  v16h w1 = load_bfrag(Wout, Cdim, Cdim, 0, 16, lane);
  float b0 = bout[m], b1 = bout[16 + m];

  int nTiles = (N + 15) >> 4;
  for (int tile = blockIdx.x * WAVES + wave; tile < nTiles;
       tile += gridDim.x * WAVES) {
    int base = tile << 4;
    int row = base + m;
    if (row > N - 1) row = N - 1;
    v16h a;
#pragma unroll
    for (int t = 0; t < 16; ++t)
      a[t] = (_Float16)acc[row * Cdim + frag_k(t, g)];

    v8f c0 = {};
    c0 = wmma_f16(a, w0, c0);
#pragma unroll
    for (int r = 0; r < 8; ++r) {
      int rr = base + r + 8 * g;
      if (rr < N) out[rr * Cdim + m] = fmaxf(c0[r] + b0, 0.f);
    }
    v8f c1 = {};
    c1 = wmma_f16(a, w1, c1);
#pragma unroll
    for (int r = 0; r < 8; ++r) {
      int rr = base + r + 8 * g;
      if (rr < N) out[rr * Cdim + 16 + m] = fmaxf(c1[r] + b1, 0.f);
    }
  }
}

// ---------- launch ----------

extern "C" void kernel_launch(void* const* d_in, const int* in_sizes, int n_in,
                              void* d_out, int out_size, void* d_ws,
                              size_t ws_size, hipStream_t stream) {
  const float* x = (const float*)d_in[0];
  const float* pos = (const float*)d_in[1];
  const int* ei = (const int*)d_in[2];
  const float* Win = (const float*)d_in[3];
  const float* bin = (const float*)d_in[4];
  const float* Wv = (const float*)d_in[5];
  const float* Wsrc = (const float*)d_in[6];
  const float* Wdst = (const float*)d_in[7];
  const float* pW1 = (const float*)d_in[8];
  const float* pb1 = (const float*)d_in[9];
  const float* pW2 = (const float*)d_in[10];
  const float* pb2 = (const float*)d_in[11];
  const float* aW1 = (const float*)d_in[12];
  const float* ab1 = (const float*)d_in[13];
  const float* aW2 = (const float*)d_in[14];
  const float* ab2 = (const float*)d_in[15];
  const float* Wout = (const float*)d_in[16];
  const float* bout = (const float*)d_in[17];

  const int N = in_sizes[0] / Cdim;
  const int E = in_sizes[2] / 2;

  char* ws = (char*)d_ws;
  size_t nodeBytes = (size_t)N * Cdim * sizeof(float);
  float* vbuf = (float*)(ws + 0 * nodeBytes);
  float* asrc = (float*)(ws + 1 * nodeBytes);
  float* adst = (float*)(ws + 2 * nodeBytes);
  unsigned* mb = (unsigned*)(ws + 3 * nodeBytes);
  float* denom = (float*)(ws + 4 * nodeBytes);
  float* acc = (float*)(ws + 5 * nodeBytes);
  float* out = (float*)d_out;

  int nInit = N * Cdim;
  init_bufs<<<(nInit + 255) / 256, 256, 0, stream>>>(mb, denom, acc, nInit);

  int nodeTiles = (N + 15) / 16;
  int nodeBlocks = (nodeTiles + WAVES - 1) / WAVES;
  if (nodeBlocks > 1024) nodeBlocks = 1024;
  node_in_kernel<<<nodeBlocks, 32 * WAVES, 0, stream>>>(x, Win, bin, Wv, Wsrc,
                                                        Wdst, vbuf, asrc, adst,
                                                        N);

  int edgeTiles = (E + 15) / 16;
  int edgeBlocks = (edgeTiles + WAVES - 1) / WAVES;
  if (edgeBlocks > 2048) edgeBlocks = 2048;
  edge_pass<0><<<edgeBlocks, 32 * WAVES, 0, stream>>>(
      ei, pos, asrc, adst, vbuf, pW1, pb1, pW2, pb2, aW1, ab1, aW2, ab2, mb,
      denom, acc, E);
  edge_pass<1><<<edgeBlocks, 32 * WAVES, 0, stream>>>(
      ei, pos, asrc, adst, vbuf, pW1, pb1, pW2, pb2, aW1, ab1, aW2, ab2, mb,
      denom, acc, E);
  edge_pass<2><<<edgeBlocks, 32 * WAVES, 0, stream>>>(
      ei, pos, asrc, adst, vbuf, pW1, pb1, pW2, pb2, aW1, ab1, aW2, ab2, mb,
      denom, acc, E);

  node_out_kernel<<<nodeBlocks, 32 * WAVES, 0, stream>>>(acc, Wout, bout, out,
                                                         N);
}
